// GCNv2_13116830122344
// MI455X (gfx1250) — matrix operand order repaired and checked by reading, not compiled, not run on hardware
//
#include <hip/hip_runtime.h>

// ---------------------------------------------------------------------------
// GCN2Conv forward for MI455X (gfx1250, wave32, WMMA).
// ---------------------------------------------------------------------------

#define NN   100000
#define NE   600000
#define DD   128
#define NG   200
#define NLAY 5
#define NOUT 10
#define ALPHA_C 0.1f
#define BETA_C  0.0077821404f      // log(1/128 + 1)
#define BN_EPS_C 1e-5f

typedef __attribute__((ext_vector_type(16))) __bf16 bf16x16;
typedef __attribute__((ext_vector_type(8)))  __bf16 bf16x8;
typedef __attribute__((ext_vector_type(8)))  float  f32x8;

__device__ __forceinline__ void atomicAddF(float* p, float v) {
    __hip_atomic_fetch_add(p, v, __ATOMIC_RELAXED, __HIP_MEMORY_SCOPE_AGENT);
}

// ------------------------------ utility -----------------------------------
__global__ void zero_kernel(float* __restrict__ p, int n) {
    int i = blockIdx.x * 256 + threadIdx.x;
    if (i < n) p[i] = 0.0f;
}

__global__ void degree_kernel(const int* __restrict__ src, const int* __restrict__ dst,
                              float* __restrict__ degO, float* __restrict__ degI, int nE) {
    int e = blockIdx.x * 256 + threadIdx.x;
    if (e < nE) {
        atomicAddF(&degO[src[e]], 1.0f);
        atomicAddF(&degI[dst[e]], 1.0f);
    }
}

// in-place: deg -> rsqrt(max(deg,1))
__global__ void normfin_kernel(float* __restrict__ p, int n) {
    int i = blockIdx.x * 256 + threadIdx.x;
    if (i < n) p[i] = rsqrtf(fmaxf(p[i], 1.0f));
}

// gcn_w[l][k][n] (f32) -> Wb[l][n][k] (bf16, transposed for contiguous B frags)
__global__ void wconv_kernel(const float* __restrict__ gcn_w, __bf16* __restrict__ Wb, int n) {
    int o = blockIdx.x * 256 + threadIdx.x;
    if (o >= n) return;
    int l = o >> 14;            // / 16384
    int r = o & 16383;
    int nn = r >> 7;            // output col
    int k  = r & 127;           // input dim
    Wb[o] = (__bf16)gcn_w[(l << 14) + (k << 7) + nn];
}

// ------------------------------ SpMM --------------------------------------
// one wave32 per edge, 4 channels per lane (float4)
__global__ __launch_bounds__(256)
void scatter_kernel(const float* __restrict__ hin, const int* __restrict__ src,
                    const int* __restrict__ dst, const float* __restrict__ norm_out,
                    float* __restrict__ agg, int nE) {
    int e    = blockIdx.x * 8 + (threadIdx.x >> 5);
    int lane = threadIdx.x & 31;
    if (e >= nE) return;
    int s = src[e], d = dst[e];
    float no = norm_out[s];
    const float4 v = *(const float4*)(hin + s * DD + lane * 4);
    float* pd = agg + d * DD + lane * 4;
    atomicAddF(pd + 0, v.x * no);
    atomicAddF(pd + 1, v.y * no);
    atomicAddF(pd + 2, v.z * no);
    atomicAddF(pd + 3, v.w * no);
}

// ------------------------- fused residual + GEMM ---------------------------
// rst = (1-a)*agg*norm_in + a*h ; out = (1-b)*rst + b*(rst @ W) + bias
// Block: 256 threads = 8 waves. M-tile 64, each wave owns one 16-col N strip,
// computes 4 stacked 16x16 tiles with v_wmma_f32_16x16x32_bf16 (K loop of 4).
#define MTILE   64
#define SAPITCH 136   // halfs; 272B rows -> bank spread
#define SWPITCH 136

__global__ __launch_bounds__(256)
void gemm_kernel(const float* __restrict__ agg, const float* __restrict__ hin,
                 const float* __restrict__ norm_in, const __bf16* __restrict__ Wb,
                 const float* __restrict__ bias, float* __restrict__ out, int nrows) {
    __shared__ __bf16 sW[DD * SWPITCH];      // W^T : [n][k]
    __shared__ __bf16 sA[MTILE * SAPITCH];   // rst tile (bf16) : [m][k]

    const int tid  = threadIdx.x;
    const int row0 = blockIdx.x * MTILE;

    // stage W^T (bf16, already transposed in global)
    for (int i = tid; i < DD * DD; i += 256)
        sW[(i >> 7) * SWPITCH + (i & 127)] = Wb[i];

    // stage A = rst (f32 math, store bf16)
    for (int i = tid; i < MTILE * DD; i += 256) {
        int r = i >> 7, c = i & 127;
        int row = row0 + r;
        float v = 0.0f;
        if (row < nrows) {
            int g = row * DD + c;
            v = (1.0f - ALPHA_C) * agg[g] * norm_in[row] + ALPHA_C * hin[g];
        }
        sA[r * SAPITCH + c] = (__bf16)v;
    }
    __syncthreads();

    const int wave  = tid >> 5;     // 0..7 -> N strip
    const int lane  = tid & 31;
    const int lhalf = lane >> 4;    // 0/1
    const int l16   = lane & 15;
    const int n0    = wave * 16;

    f32x8 acc[4];
    for (int m = 0; m < 4; ++m)
        for (int j = 0; j < 8; ++j) acc[m][j] = 0.0f;

    for (int kb = 0; kb < 4; ++kb) {
        // B fragment: lane holds col n0+l16, K = kb*32 + lhalf*16 + (0..15), contiguous
        const __bf16* pb = &sW[(n0 + l16) * SWPITCH + kb * 32 + lhalf * 16];
        bf16x8 b_lo = *(const bf16x8*)pb;
        bf16x8 b_hi = *(const bf16x8*)(pb + 8);
        bf16x16 bfrag;
        for (int j = 0; j < 8; ++j) { bfrag[j] = b_lo[j]; bfrag[j + 8] = b_hi[j]; }

        for (int m = 0; m < 4; ++m) {
            // A fragment: row m*16+l16, K = kb*32 + lhalf*16 + (0..15), contiguous
            const __bf16* pa = &sA[(m * 16 + l16) * SAPITCH + kb * 32 + lhalf * 16];
            bf16x8 a_lo = *(const bf16x8*)pa;
            bf16x8 a_hi = *(const bf16x8*)(pa + 8);
            bf16x16 afrag;
            for (int j = 0; j < 8; ++j) { afrag[j] = a_lo[j]; afrag[j + 8] = a_hi[j]; }

            acc[m] = __builtin_amdgcn_wmma_f32_16x16x32_bf16(
                false, afrag, false, bfrag, (short)0, acc[m], false, false);
        }
    }

    // epilogue: recompute rst in f32 (exact), combine, store (out aliases agg: safe,
    // each lane reads/writes only its own element after all staging reads are done)
    for (int m = 0; m < 4; ++m) {
        for (int v = 0; v < 8; ++v) {
            int rl  = m * 16 + v + lhalf * 8;   // C layout: VGPR v -> M=v (+8 for lanes 16-31)
            int row = row0 + rl;
            int col = n0 + l16;
            if (row < nrows) {
                int g = row * DD + col;
                float rv = (1.0f - ALPHA_C) * agg[g] * norm_in[row] + ALPHA_C * hin[g];
                out[g] = (1.0f - BETA_C) * rv + BETA_C * acc[m][v] + bias[col];
            }
        }
    }
}

// ------------------------------ batchnorm ----------------------------------
__global__ __launch_bounds__(128)
void bnstats_kernel(const float* __restrict__ rst, float* __restrict__ stats, int nrows) {
    int c  = threadIdx.x;
    int r0 = blockIdx.x * 64;
    float s = 0.0f, s2 = 0.0f;
    for (int r = 0; r < 64; ++r) {
        int row = r0 + r;
        if (row < nrows) {
            float v = rst[row * DD + c];
            s += v; s2 += v * v;
        }
    }
    atomicAddF(&stats[c], s);
    atomicAddF(&stats[DD + c], s2);
}

__global__ void bnfin_kernel(const float* __restrict__ stats, const float* __restrict__ bn_g,
                             const float* __restrict__ bn_b, float* __restrict__ ss, float inv_n) {
    int c = threadIdx.x;
    float mu  = stats[c] * inv_n;
    float var = stats[DD + c] * inv_n - mu * mu;   // biased variance
    float sc  = bn_g[c] * rsqrtf(var + BN_EPS_C);
    ss[c]      = sc;
    ss[DD + c] = bn_b[c] - mu * sc;
}

__global__ __launch_bounds__(256)
void bnrelu_pool_kernel(const float* __restrict__ rst, const float* __restrict__ ss,
                        const int* __restrict__ graph_id, float* __restrict__ h,
                        float* __restrict__ pooled_l, int nrows) {
    int i = blockIdx.x * 256 + threadIdx.x;
    if (i >= nrows * DD) return;
    int row = i >> 7, c = i & 127;
    float v = fmaxf(rst[i] * ss[c] + ss[DD + c], 0.0f);
    h[i] = v;
    atomicAddF(&pooled_l[graph_id[row] * DD + c], v);
}

__global__ __launch_bounds__(256)
void pool0_kernel(const float* __restrict__ feat, const int* __restrict__ graph_id,
                  float* __restrict__ pooled0, int nrows) {
    int i = blockIdx.x * 256 + threadIdx.x;
    if (i >= nrows * DD) return;
    int row = i >> 7, c = i & 127;
    atomicAddF(&pooled0[graph_id[row] * DD + c], feat[i]);
}

// ------------------------------ head ---------------------------------------
// one block per graph: 6 tiny 128x10 projections, log_softmax, mean pooled
__global__ __launch_bounds__(128)
void head_kernel(const float* __restrict__ pooled, const float* __restrict__ lin_w,
                 const float* __restrict__ lin_b, float* __restrict__ out) {
    __shared__ float sc[16];
    int g = blockIdx.x, t = threadIdx.x;
    if (t < NOUT) {
        float s = 0.0f;
        for (int l = 0; l < NLAY + 1; ++l) {
            const float* p = pooled + (size_t)(l * NG + g) * DD;
            const float* w = lin_w + l * DD * NOUT + t;
            float a = 0.0f;
            for (int d = 0; d < DD; ++d) a += p[d] * w[d * NOUT];
            s += a + lin_b[l * NOUT + t];
        }
        sc[t] = s;
    }
    __syncthreads();
    if (t < NOUT) {
        float m = -INFINITY;
        for (int o = 0; o < NOUT; ++o) m = fmaxf(m, sc[o]);
        float se = 0.0f;
        for (int o = 0; o < NOUT; ++o) se += expf(sc[o] - m);
        out[g * NOUT + t] = sc[t] - m - logf(se);
    }
    float mp = 0.0f;
    for (int l = 1; l < NLAY + 1; ++l) mp += pooled[(size_t)(l * NG + g) * DD + t];
    out[NG * NOUT + g * DD + t] = mp * 0.2f;
}

// ------------------------------ launcher -----------------------------------
extern "C" void kernel_launch(void* const* d_in, const int* in_sizes, int n_in,
                              void* d_out, int out_size, void* d_ws, size_t ws_size,
                              hipStream_t stream) {
    const float* feat     = (const float*)d_in[0];
    const int*   src      = (const int*)  d_in[1];
    const int*   dst      = (const int*)  d_in[2];
    const int*   graph_id = (const int*)  d_in[3];
    const float* gcn_w    = (const float*)d_in[4];
    const float* gcn_b    = (const float*)d_in[5];
    const float* bn_g     = (const float*)d_in[6];
    const float* bn_b     = (const float*)d_in[7];
    const float* lin_w    = (const float*)d_in[8];
    const float* lin_b    = (const float*)d_in[9];
    float*       outp     = (float*)d_out;

    // workspace layout (256B aligned chunks)
    char*  base = (char*)d_ws;
    size_t off  = 0;
    auto alloc = [&](size_t bytes) -> char* {
        char* p = base + off;
        off = (off + bytes + 255) & ~(size_t)255;
        return p;
    };
    float*  norms  = (float*)alloc((size_t)2 * NN * 4);          // deg/norm out|in
    float*  h      = (float*)alloc((size_t)NN * DD * 4);
    float*  agg    = (float*)alloc((size_t)NN * DD * 4);         // agg, then rst in place
    float*  stats  = (float*)alloc(2 * DD * 4);
    float*  ss     = (float*)alloc(2 * DD * 4);
    float*  pooled = (float*)alloc((size_t)(NLAY + 1) * NG * DD * 4);
    __bf16* Wb     = (__bf16*)alloc((size_t)NLAY * DD * DD * 2); // W^T bf16
    float* norm_out = norms;
    float* norm_in  = norms + NN;

    const int ND = NN * DD;
    auto cdiv = [](int a, int b) { return (a + b - 1) / b; };

    // init + degrees + weight conversion + pooled[0]
    zero_kernel<<<cdiv(2 * NN, 256), 256, 0, stream>>>(norms, 2 * NN);
    zero_kernel<<<cdiv((NLAY + 1) * NG * DD, 256), 256, 0, stream>>>(pooled, (NLAY + 1) * NG * DD);
    degree_kernel<<<cdiv(NE, 256), 256, 0, stream>>>(src, dst, norm_out, norm_in, NE);
    normfin_kernel<<<cdiv(2 * NN, 256), 256, 0, stream>>>(norms, 2 * NN);
    wconv_kernel<<<cdiv(NLAY * DD * DD, 256), 256, 0, stream>>>(gcn_w, Wb, NLAY * DD * DD);
    pool0_kernel<<<cdiv(ND, 256), 256, 0, stream>>>(feat, graph_id, pooled, NN);

    for (int l = 0; l < NLAY; ++l) {
        const float* hin = (l == 0) ? feat : h;
        zero_kernel<<<cdiv(ND, 256), 256, 0, stream>>>(agg, ND);
        zero_kernel<<<1, 256, 0, stream>>>(stats, 2 * DD);
        scatter_kernel<<<cdiv(NE, 8), 256, 0, stream>>>(hin, src, dst, norm_out, agg, NE);
        gemm_kernel<<<cdiv(NN, MTILE), 256, 0, stream>>>(
            agg, hin, norm_in, Wb + (size_t)l * DD * DD, gcn_b + l * DD, agg, NN);
        bnstats_kernel<<<cdiv(NN, 64), 128, 0, stream>>>(agg, stats, NN);
        bnfin_kernel<<<1, 128, 0, stream>>>(stats, bn_g + l * DD, bn_b + l * DD, ss, 1.0f / NN);
        bnrelu_pool_kernel<<<cdiv(ND, 256), 256, 0, stream>>>(
            agg, ss, graph_id, h, pooled + (size_t)(l + 1) * NG * DD, NN);
    }

    head_kernel<<<NG, 128, 0, stream>>>(pooled, lin_w, lin_b, outp);
}
